// GCN_77352361001233
// MI455X (gfx1250) — compile-verified, matrix operand order checked by reading, not run on hardware
//
#include <hip/hip_runtime.h>
#include <hip/hip_bf16.h>

#define Nn   100000
#define Ee   1600000
#define Gg   64
#define FDIM 128
#define CLSn 10

typedef __attribute__((ext_vector_type(2)))  float  v2f;
typedef __attribute__((ext_vector_type(8)))  float  v8f;
typedef __attribute__((ext_vector_type(16))) __bf16 v16bf;

// ---------------------------------------------------------------------------
// Edge preprocessing: deg (src-side), dinv, w[e] = -dinv[src]*dinv[dst]
// ---------------------------------------------------------------------------
__global__ void __launch_bounds__(256) deg_kernel(const int* __restrict__ src,
                                                  float* __restrict__ deg) {
    int e = blockIdx.x * blockDim.x + threadIdx.x;
    if (e < Ee) atomicAdd(deg + src[e], 1.0f);
}

__global__ void __launch_bounds__(256) dinv_kernel(float* __restrict__ deg) {
    int i = blockIdx.x * blockDim.x + threadIdx.x;
    if (i < Nn) {
        float d = deg[i];
        deg[i] = (d > 0.0f) ? rsqrtf(fmaxf(d, 1.0f)) : 0.0f;
    }
}

__global__ void __launch_bounds__(256) edgew_kernel(const int* __restrict__ src,
                                                    const int* __restrict__ dst,
                                                    const float* __restrict__ dinv,
                                                    float* __restrict__ w) {
    int e = blockIdx.x * blockDim.x + threadIdx.x;
    if (e < Ee) w[e] = -dinv[src[e]] * dinv[dst[e]];
}

// ---------------------------------------------------------------------------
// SpMM scatter: out[dst[e]] += w[e] * x[src[e]]  (one wave per edge,
// lane owns 4 contiguous features; gather + scatter are L2-resident).
// Edge index is wave-uniform -> readfirstlane so src/dst/w become scalar
// loads (KMcnt path) instead of 3 broadcast VMEM loads per wave.
// ---------------------------------------------------------------------------
__global__ void __launch_bounds__(256)
spmm_scatter(const float* __restrict__ x, const int* __restrict__ src,
             const int* __restrict__ dst, const float* __restrict__ w,
             float* __restrict__ out) {
    int e = __builtin_amdgcn_readfirstlane(blockIdx.x * 8 + (threadIdx.x >> 5));
    if (e >= Ee) return;
    int lane = threadIdx.x & 31;
    int s = src[e], d = dst[e];
    float we = w[e];
    float4 xv = ((const float4*)(x + (size_t)s * FDIM))[lane];
    float* o = out + (size_t)d * FDIM + lane * 4;
    atomicAdd(o + 0, we * xv.x);
    atomicAdd(o + 1, we * xv.y);
    atomicAdd(o + 2, we * xv.z);
    atomicAdd(o + 3, we * xv.w);
}

// t2 = 2*t2 - t0  (Chebyshev recurrence)
__global__ void __launch_bounds__(256)
cheb_t2(float4* __restrict__ t2, const float4* __restrict__ t0, int n4) {
    int i = blockIdx.x * blockDim.x + threadIdx.x;
    if (i < n4) {
        float4 a = t2[i], b = t0[i];
        a.x = 2.0f * a.x - b.x;
        a.y = 2.0f * a.y - b.y;
        a.z = 2.0f * a.z - b.z;
        a.w = 2.0f * a.w - b.w;
        t2[i] = a;
    }
}

// ---------------------------------------------------------------------------
// WMMA GEMM: out[m][n] (+)= A[m][:] @ W[:][n], K = 128, fused bias / ReLU.
// Block = 128 threads = 4 waves; each wave computes a 16x16 tile,
// block covers 64 rows x 16 cols. W tile (128x16) staged in LDS.
//
// Boundary handling: A row m only contributes to output row m, and
// out-of-range rows are discarded at the store — out-of-range A reads are
// CLAMPED to row nrows-1 (unconditional b64 loads, no exec-mask churn).
//
// K loop runs in 32-wide chunks: 8 A frags + 8 B frags preloaded into
// registers, then 8 back-to-back WMMAs — gives the LDS pipeline an
// 8-deep ds clause to hide latency (first WMMA waits at dscnt<=7).
// ---------------------------------------------------------------------------
__global__ void __launch_bounds__(128)
cheb_gemm(const float* __restrict__ A, const float* __restrict__ W,
          const float* __restrict__ bias, float* __restrict__ out,
          int nrows, int accumulate, int relu) {
    __shared__ float Wlds[FDIM * 16];

    const int tid   = threadIdx.x;
    const int wave  = tid >> 5;
    const int lane  = tid & 31;
    const int half  = lane >> 4;   // 0 or 1
    const int idx16 = lane & 15;
    const int n0    = blockIdx.y * 16;
    const int row0  = blockIdx.x * 64 + wave * 16;

    // A fragment row: lanes 0-15 and 16-31 both map M = lane&15 (clamped).
    const int mA = row0 + idx16;
    const float* Arow = A + (size_t)(mA < nrows ? mA : (nrows - 1)) * FDIM;
    __builtin_prefetch(Arow, 0, 0);               // global_prefetch_b8

    // Stage W[k][n0..n0+15] tile: thread tid loads row k = tid (16 floats).
    {
        const float4* Wr = (const float4*)(W + (size_t)tid * FDIM + n0);
        float4* Ld = (float4*)(Wlds + tid * 16);
        Ld[0] = Wr[0]; Ld[1] = Wr[1]; Ld[2] = Wr[2]; Ld[3] = Wr[3];
    }
    __syncthreads();

    // C/D layout: VGPR j -> M = row0 + j + 8*half, N = n0 + idx16.
    v8f c;
    if (accumulate) {
#pragma unroll
        for (int j = 0; j < 8; ++j) {
            int m  = row0 + j + half * 8;
            int mc = m < nrows ? m : (nrows - 1);
            c[j] = out[(size_t)mc * FDIM + n0 + idx16];
        }
    } else {
        float bval = bias ? bias[n0 + idx16] : 0.0f;
#pragma unroll
        for (int j = 0; j < 8; ++j) c[j] = bval;
    }

#if __has_builtin(__builtin_amdgcn_wmma_f32_16x16x4_f32)
#pragma unroll
    for (int kc = 0; kc < FDIM; kc += 32) {
        v2f a[8], b[8];
#pragma unroll
        for (int i = 0; i < 8; ++i) {
            const int ka = kc + 4 * i + half * 2;   // lanes 16-31: K = +2,+3
            a[i] = *(const v2f*)(Arow + ka);        // unconditional b64 load
            b[i][0] = Wlds[ka * 16 + idx16];
            b[i][1] = Wlds[(ka + 1) * 16 + idx16];
        }
#pragma unroll
        for (int i = 0; i < 8; ++i)
            c = __builtin_amdgcn_wmma_f32_16x16x4_f32(
                    false, a[i], false, b[i], (short)0, c, false, false);
    }
#else
    // bf16 fallback (codegen-confirmed builtin); K step 32.
#pragma unroll
    for (int k0 = 0; k0 < FDIM; k0 += 32) {
        v16bf a, b;
#pragma unroll
        for (int v = 0; v < 8; ++v) {
            int kb = k0 + (v & 3) * 2 + half * 8 + (v >> 2) * 16;
            a[2 * v]     = (__bf16)Arow[kb];
            a[2 * v + 1] = (__bf16)Arow[kb + 1];
            b[2 * v]     = (__bf16)Wlds[kb * 16 + idx16];
            b[2 * v + 1] = (__bf16)Wlds[(kb + 1) * 16 + idx16];
        }
        c = __builtin_amdgcn_wmma_f32_16x16x32_bf16(
                false, a, false, b, (short)0, c, false, false);
    }
#endif

#pragma unroll
    for (int j = 0; j < 8; ++j) {
        int m = row0 + j + half * 8;
        if (m < nrows) {
            float v = c[j];
            if (relu) v = fmaxf(v, 0.0f);
            out[(size_t)m * FDIM + n0 + idx16] = v;
        }
    }
}

// ---------------------------------------------------------------------------
// Pooling + head
// ---------------------------------------------------------------------------
__global__ void __launch_bounds__(256) pool_cnt(const int* __restrict__ batch,
                                                float* __restrict__ cnt) {
    int i = blockIdx.x * blockDim.x + threadIdx.x;
    if (i < Nn) atomicAdd(cnt + batch[i], 1.0f);
}

__global__ void __launch_bounds__(256)
pool_sum(const float* __restrict__ h, const int* __restrict__ batch,
         float* __restrict__ pooled) {
    int n = __builtin_amdgcn_readfirstlane(blockIdx.x * 8 + (threadIdx.x >> 5));
    if (n >= Nn) return;
    int lane = threadIdx.x & 31;
    int g = batch[n];
    float4 v = ((const float4*)(h + (size_t)n * FDIM))[lane];
    float* p = pooled + (size_t)g * FDIM + lane * 4;
    atomicAdd(p + 0, v.x);
    atomicAdd(p + 1, v.y);
    atomicAdd(p + 2, v.z);
    atomicAdd(p + 3, v.w);
}

__global__ void __launch_bounds__(256)
head_kernel(const float* __restrict__ pooled, const float* __restrict__ cnt,
            const float* __restrict__ Wl, const float* __restrict__ bl,
            float* __restrict__ out) {
    int t = blockIdx.x * blockDim.x + threadIdx.x;
    if (t >= Gg * CLSn) return;
    int g = t / CLSn, cc = t % CLSn;
    float inv = 1.0f / fmaxf(cnt[g], 1.0f);
    float acc = bl[cc];
    for (int f = 0; f < FDIM; ++f)
        acc += pooled[(size_t)g * FDIM + f] * inv * Wl[f * CLSn + cc];
    out[t] = acc;
}

// ---------------------------------------------------------------------------
extern "C" void kernel_launch(void* const* d_in, const int* in_sizes, int n_in,
                              void* d_out, int out_size, void* d_ws, size_t ws_size,
                              hipStream_t stream) {
    const float* x     = (const float*)d_in[0];
    const int*   ei    = (const int*)d_in[1];
    const int*   batch = (const int*)d_in[2];
    const float* W1 = (const float*)d_in[3];
    const float* b1 = (const float*)d_in[4];
    const float* W2 = (const float*)d_in[5];
    const float* b2 = (const float*)d_in[6];
    const float* W3 = (const float*)d_in[7];
    const float* b3 = (const float*)d_in[8];
    const float* Wl = (const float*)d_in[9];
    const float* bl = (const float*)d_in[10];
    const int* src = ei;
    const int* dst = ei + Ee;

    float* ws = (float*)d_ws;
    const size_t FB = (size_t)Nn * FDIM;
    float* F0     = ws;
    float* F1     = ws + FB;
    float* F2     = ws + 2 * FB;
    float* F3     = ws + 3 * FB;
    float* wbuf   = ws + 4 * FB;
    float* deg    = wbuf + Ee;
    float* pooled = deg + Nn;
    float* cnt    = pooled + (size_t)Gg * FDIM;

    // edge weights
    hipMemsetAsync(deg, 0, Nn * sizeof(float), stream);
    deg_kernel  <<<(Ee + 255) / 256, 256, 0, stream>>>(src, deg);
    dinv_kernel <<<(Nn + 255) / 256, 256, 0, stream>>>(deg);
    edgew_kernel<<<(Ee + 255) / 256, 256, 0, stream>>>(src, dst, deg, wbuf);

    dim3 ggrid((Nn + 63) / 64, FDIM / 16);
    const int spmm_blocks = (Ee + 7) / 8;     // 8 edge-waves per 256-thr block
    const int t2_blocks   = (int)((FB / 4 + 255) / 256);

    auto layer = [&](const float* in, float* t1, float* t2, float* outp,
                     const float* W, const float* b, int relu) {
        // out = T0 @ W0 + b
        cheb_gemm<<<ggrid, 128, 0, stream>>>(in, W, b, outp, Nn, 0, 0);
        // T1 = prop(T0)
        hipMemsetAsync(t1, 0, FB * sizeof(float), stream);
        spmm_scatter<<<spmm_blocks, 256, 0, stream>>>(in, src, dst, wbuf, t1);
        // out += T1 @ W1
        cheb_gemm<<<ggrid, 128, 0, stream>>>(t1, W + FDIM * FDIM, nullptr, outp, Nn, 1, 0);
        // T2 = 2*prop(T1) - T0
        hipMemsetAsync(t2, 0, FB * sizeof(float), stream);
        spmm_scatter<<<spmm_blocks, 256, 0, stream>>>(t1, src, dst, wbuf, t2);
        cheb_t2<<<t2_blocks, 256, 0, stream>>>((float4*)t2, (const float4*)in, (int)(FB / 4));
        // out += T2 @ W2  (+ReLU on hidden layers)
        cheb_gemm<<<ggrid, 128, 0, stream>>>(t2, W + 2 * FDIM * FDIM, nullptr, outp, Nn, 1, relu);
    };

    layer(x,  F0, F1, F2, W1, b1, 1);
    layer(F2, F0, F1, F3, W2, b2, 1);
    layer(F3, F0, F1, F2, W3, b3, 0);

    // mean pool + linear head
    hipMemsetAsync(pooled, 0, (Gg * FDIM + Gg) * sizeof(float), stream);
    pool_cnt<<<(Nn + 255) / 256, 256, 0, stream>>>(batch, cnt);
    pool_sum<<<(Nn + 7) / 8, 256, 0, stream>>>(F2, batch, pooled);
    head_kernel<<<(Gg * CLSn + 255) / 256, 256, 0, stream>>>(pooled, cnt, Wl, bl, (float*)d_out);
}
